// MultiSourceLSTM_6949257085186
// MI455X (gfx1250) — matrix operand order repaired and checked by reading, not compiled, not run on hardware
//
#include <hip/hip_runtime.h>
#include <hip/hip_bf16.h>

typedef __attribute__((ext_vector_type(16))) _Float16 v16h;
typedef __attribute__((ext_vector_type(8)))  float    v8f;

#define B_   1024
#define T_   512
#define F_   40
#define G_   5
#define FG_  8
#define H_   13
#define HID_ 65

// ---- fast activations: keep the serial recurrence path short ------------
#if __has_builtin(__builtin_amdgcn_tanhf)
__device__ __forceinline__ float ftanh(float x) { return __builtin_amdgcn_tanhf(x); }
__device__ __forceinline__ float fsig(float x)  { return fmaf(0.5f, __builtin_amdgcn_tanhf(0.5f * x), 0.5f); }
#else
#if __has_builtin(__builtin_amdgcn_rcpf)
__device__ __forceinline__ float frcp_fast(float x) { return __builtin_amdgcn_rcpf(x); }
#else
__device__ __forceinline__ float frcp_fast(float x) { return 1.0f / x; }
#endif
__device__ __forceinline__ float fsig(float x)  { return frcp_fast(1.0f + __expf(-x)); }
__device__ __forceinline__ float ftanh(float x) { return 1.0f - 2.0f * frcp_fast(__expf(2.0f * x) + 1.0f); }
#endif

// One wave (32 threads) per (16-batch tile, group). Runs the full T=512
// recurrence with one fused WMMA per gate-tile per step:
//   gates = [x_t | h] (16x32 f16)  x  [W_ih^T ; W_hh^T] (32x16 f16)  +  bias(C)
// Gate columns remapped to tiles {i,f,g,o} each padded 13->16 so the cell
// update is purely per-lane (no cross-lane ops in the time loop except the
// 512-byte LDS transpose bounce of h).
__global__ __launch_bounds__(32) void lstm_wmma_kernel(
    const float* __restrict__ x,     // [B,T,F]
    const float* __restrict__ W_ih,  // [G,4H,FG]
    const float* __restrict__ W_hh,  // [G,4H,H]
    const float* __restrict__ b_ih,  // [G,4H]
    const float* __restrict__ b_hh,  // [G,4H]
    float* __restrict__ combined)    // [B,HID]
{
    __shared__ _Float16 hbuf[16][16];   // h in [m][j] layout, row stride 32B

    const int lane  = threadIdx.x;      // wave32
    const int n     = lane & 15;        // N column / A row M
    const int khalf = lane >> 4;        // which K-half this lane holds
    const int wid   = blockIdx.x;
    const int g     = wid % G_;
    const int b0    = (wid / G_) * 16;

    // zero LDS h buffer (also provides the permanent zero padding j=13..15)
    for (int i = lane; i < 256; i += 32) ((_Float16*)hbuf)[i] = (_Float16)0.0f;
    __syncthreads();

    // ---- Build B tiles (32x16 f16) and bias, once ----------------------
    // B layout: lane L, element e -> K = e<8 ? (L>=16)*8 + e : 16 + (L>=16)*8 + (e-8), N = L&15.
    // Column n of tile tg maps to original gate row tg*13+n (n<13), else 0.
    v16h  bm[4];
    float bias[4];
#pragma unroll
    for (int tg = 0; tg < 4; ++tg) {
        v16h bt;
#pragma unroll
        for (int e = 0; e < 16; ++e) {
            const int k = (e < 8) ? khalf * 8 + e : 16 + khalf * 8 + (e - 8);
            float w = 0.0f;
            if (n < H_) {
                const int row = g * 4 * H_ + tg * H_ + n;
                if (k < FG_)           w = W_ih[row * FG_ + k];
                else if (k < FG_ + H_) w = W_hh[row * H_  + (k - FG_)];
            }
            bt[e] = (_Float16)w;
        }
        bm[tg] = bt;
        float bb = 0.0f;
        if (n < H_) {
            const int row = g * 4 * H_ + tg * H_ + n;
            bb = b_ih[row] + b_hh[row];
        }
        bias[tg] = bb;
    }

    // ---- x stream: lanes mirror rows (lanes 16..31 duplicate, stays convergent)
    const float* xp = x + (size_t)(b0 + n) * T_ * F_ + g * FG_;
    float4 xa = *(const float4*)(xp);           // t = 0
    float4 xb = *(const float4*)(xp + 4);

    float cst[8] = {0,0,0,0,0,0,0,0};
    float hst[8] = {0,0,0,0,0,0,0,0};
    const bool lo = (khalf == 0);

    for (int t = 0; t < T_; ++t) {
        // ---- build A (16x32 f16): [x(K0..7) | h(K8..20) | zeros] --------
        const float xf[8] = {xa.x, xa.y, xa.z, xa.w, xb.x, xb.y, xb.z, xb.w};
        union { uint4 u4; _Float16 h8[8]; } lh;
        // lanes 0-15 need h j=8..15 (K16..23); lanes 16-31 need h j=0..7 (K8..15)
        lh.u4 = *(const uint4*)&hbuf[n][lo ? 8 : 0];

        v16h a;
#pragma unroll
        for (int e = 0; e < 8; ++e) a[e]     = lo ? (_Float16)xf[e] : lh.h8[e];
#pragma unroll
        for (int e = 0; e < 8; ++e) a[8 + e] = lo ? lh.h8[e] : (_Float16)0.0f;

        // prefetch next step's x while the WMMAs run
        float4 nxa = make_float4(0, 0, 0, 0), nxb = make_float4(0, 0, 0, 0);
        if (t + 1 < T_) {
            const float* xn = xp + (size_t)(t + 1) * F_;
            nxa = *(const float4*)(xn);
            nxb = *(const float4*)(xn + 4);
        }

        // ---- 4 independent WMMAs (i, f, g, o tiles), bias in C ----------
        v8f d[4];
#pragma unroll
        for (int tg = 0; tg < 4; ++tg) {
            v8f cc;
#pragma unroll
            for (int r = 0; r < 8; ++r) cc[r] = bias[tg];
            d[tg] = __builtin_amdgcn_wmma_f32_16x16x32_f16(
                false, a, false, bm[tg], (short)0, cc, false, false);
        }

        // ---- per-lane LSTM cell update (same m,j across the 4 tiles) ----
#pragma unroll
        for (int r = 0; r < 8; ++r) {
            const float iv = fsig(d[0][r]);
            const float fv = fsig(d[1][r]);
            const float gv = ftanh(d[2][r]);
            const float ov = fsig(d[3][r]);
            cst[r] = fv * cst[r] + iv * gv;
            hst[r] = ov * ftanh(cst[r]);
        }

        // ---- publish new h to LDS in [m][j] layout ----------------------
        if (n < H_) {
#pragma unroll
            for (int r = 0; r < 8; ++r)
                hbuf[khalf * 8 + r][n] = (_Float16)hst[r];
        }
        __syncthreads();   // order stores vs. next iteration's reads

        xa = nxa; xb = nxb;
    }

    // ---- write last hidden state: combined[b][g*13 + j] -----------------
    if (n < H_) {
#pragma unroll
        for (int r = 0; r < 8; ++r)
            combined[(size_t)(b0 + khalf * 8 + r) * HID_ + g * H_ + n] = hst[r];
    }
}

// Tiny fusion MLP: out[b] = W2 . relu(W1 . combined[b] + b1) + b2
__global__ __launch_bounds__(256) void mlp_kernel(
    const float* __restrict__ combined, const float* __restrict__ W1,
    const float* __restrict__ b1, const float* __restrict__ W2,
    const float* __restrict__ b2, float* __restrict__ out)
{
    __shared__ float sW1[32 * HID_];
    __shared__ float sb1[32];
    __shared__ float sW2[32];
    const int tid = threadIdx.x;
    for (int i = tid; i < 32 * HID_; i += 256) sW1[i] = W1[i];
    if (tid < 32) { sb1[tid] = b1[tid]; sW2[tid] = W2[tid]; }
    __syncthreads();

    const int b = blockIdx.x * 256 + tid;
    if (b < B_) {
        float comb[HID_];
#pragma unroll
        for (int k = 0; k < HID_; ++k) comb[k] = combined[(size_t)b * HID_ + k];
        float acc = b2[0];
        for (int h = 0; h < 32; ++h) {
            float s = sb1[h];
#pragma unroll
            for (int k = 0; k < HID_; ++k) s += comb[k] * sW1[h * HID_ + k];
            acc += sW2[h] * fmaxf(s, 0.0f);
        }
        out[b] = acc;
    }
}

extern "C" void kernel_launch(void* const* d_in, const int* in_sizes, int n_in,
                              void* d_out, int out_size, void* d_ws, size_t ws_size,
                              hipStream_t stream) {
    const float* x    = (const float*)d_in[0];
    const float* W_ih = (const float*)d_in[1];
    const float* W_hh = (const float*)d_in[2];
    const float* b_ih = (const float*)d_in[3];
    const float* b_hh = (const float*)d_in[4];
    const float* W1   = (const float*)d_in[5];
    const float* b1   = (const float*)d_in[6];
    const float* W2   = (const float*)d_in[7];
    const float* b2   = (const float*)d_in[8];
    float* combined   = (float*)d_ws;             // B*HID floats = 260 KiB
    float* out        = (float*)d_out;

    dim3 grid1((B_ / 16) * G_), blk1(32);         // 320 wave32 workgroups
    lstm_wmma_kernel<<<grid1, blk1, 0, stream>>>(x, W_ih, W_hh, b_ih, b_hh, combined);

    dim3 grid2((B_ + 255) / 256), blk2(256);
    mlp_kernel<<<grid2, blk2, 0, stream>>>(combined, W1, b1, W2, b2, out);
}
